// KendallDependenceMetric_77773267796369
// MI455X (gfx1250) — compile-verified, matrix operand order checked by reading, not run on hardware
//
#include <hip/hip_runtime.h>
#include <hip/hip_bf16.h>
#include <stdint.h>

typedef __attribute__((ext_vector_type(8))) int v8i;

#define TILE_J 64
#define WAVES  8
#define BLOCK_I 128  // WAVES * 16 rows per block

// ---------------------------------------------------------------------------
// Zero the histogram workspace (harness poisons ws; we must zero every call).
// ---------------------------------------------------------------------------
__global__ void __launch_bounds__(256)
kdm_zero_kernel(unsigned* __restrict__ p, int n) {
    int i = blockIdx.x * blockDim.x + threadIdx.x;
    if (i < n) p[i] = 0u;
}

// ---------------------------------------------------------------------------
// Async global -> LDS staging of one 64-row x 8-float j-tile (2 KB).
// 128 threads x 16 bytes, GVS addressing (SGPR base + VGPR offset).
// ---------------------------------------------------------------------------
__device__ __forceinline__ void kdm_stage_tile(const float* src, int jTile,
                                               unsigned ldsByteOff) {
    if (threadIdx.x < 128) {
        unsigned gOff = (unsigned)jTile * (TILE_J * 32u) + threadIdx.x * 16u;
        unsigned lOff = ldsByteOff + threadIdx.x * 16u;
        asm volatile("global_load_async_to_lds_b128 %0, %1, %2 offset:0"
                     :
                     : "v"(lOff), "v"(gOff), "s"(src)
                     : "memory");
    }
}

// ---------------------------------------------------------------------------
// Dominance-count -> histogram kernel.
// blockIdx.y selects X (0) or X_hat (1). Each wave handles a 16-row i-tile,
// loops over j in 64-wide tiles, packs dominance bits into the 16x64 IU8
// A-matrix layout, and accumulates row sums with V_WMMA_I32_16X16X64_IU8
// against an all-ones B matrix.
// ---------------------------------------------------------------------------
__global__ void __launch_bounds__(256)
kdm_dominance_hist_kernel(const float* __restrict__ X,
                          const float* __restrict__ Xh,
                          unsigned* __restrict__ h1,
                          unsigned* __restrict__ h2,
                          int N) {
    const float* src   = (blockIdx.y == 0) ? X : Xh;
    unsigned*    hist  = (blockIdx.y == 0) ? h1 : h2;

    __shared__ float tile[2][TILE_J * 8];  // double-buffered, 4 KB

    const int tid  = threadIdx.x;
    const int lane = tid & 31;
    const int wav  = tid >> 5;         // wave32: 8 waves per block
    const int m    = lane & 15;        // M within the 16-row tile
    const int h    = (lane >> 4) & 1;  // lane-half selects K sub-blocks

    const int iBase = blockIdx.x * BLOCK_I + wav * 16;
    const int iRow  = iBase + m;

    float xi[8];
#pragma unroll
    for (int d = 0; d < 8; ++d) xi[d] = src[iRow * 8 + d];

    const unsigned lds0 = (unsigned)(uintptr_t)&tile[0][0];

    v8i acc = {0, 0, 0, 0, 0, 0, 0, 0};
    const int c1 = 0x01010101;
    v8i onesB = {c1, c1, c1, c1, c1, c1, c1, c1};

    const int nTiles = N / TILE_J;

    // Prime buffer 0.
    kdm_stage_tile(src, 0, lds0);
    asm volatile("s_wait_asynccnt 0" ::: "memory");
    __syncthreads();

    for (int t = 0; t < nTiles; ++t) {
        const int cur = t & 1;
        if (t + 1 < nTiles)
            kdm_stage_tile(src, t + 1, lds0 + (unsigned)(cur ^ 1) * (TILE_J * 32u));

        const float* buf = &tile[cur][0];

        // Pack dominance bits for this lane's 32 (i, j) pairs into the
        // 8-bit A-matrix layout: K = (v>>1)*16 + h*8 + (v&1)*4 + byte.
        v8i a;
#pragma unroll
        for (int v = 0; v < 8; ++v) {
            const int kb = ((v >> 1) << 4) + (h << 3) + ((v & 1) << 2);
            unsigned pk = 0;
#pragma unroll
            for (int b = 0; b < 4; ++b) {
                const float* xj = &buf[(kb + b) * 8];
                unsigned dgt =
                    (unsigned)(xi[0] > xj[0]) & (unsigned)(xi[1] > xj[1]) &
                    (unsigned)(xi[2] > xj[2]) & (unsigned)(xi[3] > xj[3]) &
                    (unsigned)(xi[4] > xj[4]) & (unsigned)(xi[5] > xj[5]) &
                    (unsigned)(xi[6] > xj[6]) & (unsigned)(xi[7] > xj[7]);
                pk |= dgt << (8 * b);
            }
            a[v] = (int)pk;
        }

        // D = A(16x64 u8 dominance bits) x B(64x16 ones) + C  -> row sums.
        acc = __builtin_amdgcn_wmma_i32_16x16x64_iu8(
            /*sgn_a=*/false, a, /*sgn_b=*/false, onesB, acc,
            /*reuse_a=*/false, /*reuse_b=*/false);

        asm volatile("s_wait_asynccnt 0" ::: "memory");
        __syncthreads();
    }

    // D layout: VGPR r, lanes 0-15 hold M=r; lanes 16-31 hold M=r+8.
    // Column 0 (lane m==0 of each half) carries the row counts.
    if (m == 0) {
#pragma unroll
        for (int r = 0; r < 8; ++r) {
            unsigned cnt = (unsigned)acc[r];  // 0 .. N-1
            atomicAdd(&hist[cnt], 1u);
        }
    }
}

// ---------------------------------------------------------------------------
// Finalize: result = sum_t |CDF1(t) - CDF2(t)| / ((N-1)*N)
// Single block: per-thread chunk sums, block scan, abs-accumulate, reduce.
// ---------------------------------------------------------------------------
__global__ void __launch_bounds__(256)
kdm_finalize_kernel(const unsigned* __restrict__ h1,
                    const unsigned* __restrict__ h2,
                    float* __restrict__ out, int N) {
    __shared__ int       s_sum[256];
    __shared__ long long s_acc[256];

    const int tid   = threadIdx.x;
    const int chunk = N / 256;  // 32 bins per thread
    const int base  = tid * chunk;

    int s = 0;
    for (int k = 0; k < chunk; ++k)
        s += (int)h1[base + k] - (int)h2[base + k];
    s_sum[tid] = s;
    __syncthreads();

    // Hillis-Steele inclusive scan over the 256 chunk sums.
    for (int off = 1; off < 256; off <<= 1) {
        int mine  = s_sum[tid];
        int other = (tid >= off) ? s_sum[tid - off] : 0;
        __syncthreads();
        s_sum[tid] = mine + other;
        __syncthreads();
    }
    int run = (tid == 0) ? 0 : s_sum[tid - 1];  // exclusive prefix
    __syncthreads();

    long long acc = 0;
    for (int k = 0; k < chunk; ++k) {
        run += (int)h1[base + k] - (int)h2[base + k];
        acc += (run < 0) ? (long long)(-run) : (long long)run;
    }
    s_acc[tid] = acc;
    __syncthreads();
    for (int off = 128; off > 0; off >>= 1) {
        if (tid < off) s_acc[tid] += s_acc[tid + off];
        __syncthreads();
    }
    if (tid == 0) {
        double tot = (double)s_acc[0];
        out[0] = (float)(tot / ((double)(N - 1) * (double)N));
    }
}

// ---------------------------------------------------------------------------
extern "C" void kernel_launch(void* const* d_in, const int* in_sizes, int n_in,
                              void* d_out, int out_size, void* d_ws, size_t ws_size,
                              hipStream_t stream) {
    const float* X  = (const float*)d_in[0];
    const float* Xh = (const float*)d_in[1];
    const int N = in_sizes[0] / 8;  // 8192 rows, D = 8

    unsigned* h1 = (unsigned*)d_ws;
    unsigned* h2 = h1 + N;

    const int zn = 2 * N;
    kdm_zero_kernel<<<(zn + 255) / 256, 256, 0, stream>>>(h1, zn);

    dim3 grid(N / BLOCK_I, 2, 1);
    kdm_dominance_hist_kernel<<<grid, 256, 0, stream>>>(X, Xh, h1, h2, N);

    kdm_finalize_kernel<<<1, 256, 0, stream>>>(h1, h2, (float*)d_out, N);
}
